// Decoder_335007449569
// MI455X (gfx1250) — matrix operand order repaired and checked by reading, not compiled, not run on hardware
//
#include <hip/hip_runtime.h>
#include <hip/hip_bf16.h>

typedef __attribute__((ext_vector_type(16))) _Float16 v16h;
typedef __attribute__((ext_vector_type(8)))  _Float16 v8h;
typedef __attribute__((ext_vector_type(8)))  float    v8f;

#define EMB_DIM   300
#define KP        320            // padded K, 10 x 32
#define KSTEPS    10
#define BATCH     1024
#define LEN_TOK   64
#define VOCAB     50000          // 16 * 3125 exactly
#define NTILES    3125
#define NCHUNKS   125
#define TPC       25             // tiles per chunk: 125*25 = 3125
#define MGROUPS   8              // 8 waves * 16 rows = 128 rows per WG; 1024/128 = 8
#define BROW      328            // padded LDS row stride in f16 (bank-conflict mitigation)

// ---------------------------------------------------------------------------
// f32 -> f16 conversion with K padding 300 -> 320 (zeros)
// ---------------------------------------------------------------------------
__global__ void cvt_pad_kernel(const float* __restrict__ src,
                               _Float16* __restrict__ dst, int rows) {
    size_t i = (size_t)blockIdx.x * blockDim.x + threadIdx.x;
    size_t total = (size_t)rows * KP;
    if (i >= total) return;
    int    k = (int)(i % KP);
    size_t r = i / KP;
    dst[i] = (k < EMB_DIM) ? (_Float16)src[r * EMB_DIM + k] : (_Float16)0.f;
}

// ---------------------------------------------------------------------------
// A fragment (16x32 f16, per ISA 7.12.2): lane l<16 -> row l, K{0..7,16..23};
// lane>=16 -> row l-16, K{8..15,24..31}. Loaded straight from global (tiny, L2).
// ---------------------------------------------------------------------------
__device__ __forceinline__ v16h load_fragA(const _Float16* __restrict__ A,
                                           int row0, int k0) {
    int l = threadIdx.x & 31;
    int r = l & 15;
    int koff = (l < 16) ? 0 : 8;
    const _Float16* p = A + (size_t)(row0 + r) * KP + k0 + koff;
    v8h lo = *(const v8h*)p;
    v8h hi = *(const v8h*)(p + 16);
    v16h f;
#pragma unroll
    for (int i = 0; i < 8; ++i) { f[i] = lo[i]; f[i + 8] = hi[i]; }
    return f;
}

// ---------------------------------------------------------------------------
// Fused WMMA GEMM + online-softmax partial LSE, with async global->LDS B
// staging shared across 8 waves (8 M-tiles) per workgroup.
// grid = (NCHUNKS, MGROUPS), block = 256 (8 waves).
// partials[row * NCHUNKS + chunk] = (max, sumexp) over that vocab chunk.
// ---------------------------------------------------------------------------
__global__ __launch_bounds__(256)
void lse_gemm_kernel(const _Float16* __restrict__ A,    // [1024 x 320] f16
                     const _Float16* __restrict__ E,    // [50000 x 320] f16
                     const float*    __restrict__ bias, // [50000]
                     float2*         __restrict__ partials) // [1024 x NCHUNKS]
{
    const int chunk = blockIdx.x;
    const int wave  = threadIdx.x >> 5;
    const int lane  = threadIdx.x & 31;
    const int mtile = blockIdx.y * 8 + wave;   // each wave owns a distinct M-tile
    const int row0  = mtile * 16;

    // Double-buffered B tile: 16 rows x 320 f16, padded to stride 328.
    __shared__ __align__(32) _Float16 Bbuf[2][16 * BROW];

    // Hoist all A fragments for this wave's M-tile (reused across every N tile).
    v16h afrag[KSTEPS];
#pragma unroll
    for (int kt = 0; kt < KSTEPS; ++kt)
        afrag[kt] = load_fragA(A, row0, kt * 32);

    // Per-thread async-copy pattern: tile = 1280 b64 chunks; 5 per thread.
    uint32_t ldsoff[5];
    uint64_t goff[5];
#pragma unroll
    for (int j = 0; j < 5; ++j) {
        int id = (int)threadIdx.x + j * 256;   // 0..1279
        int r  = id / 80;                      // row within tile (80 b64 per row)
        int c  = id % 80;
        ldsoff[j] = (uint32_t)(r * (BROW * 2) + c * 8);  // padded LDS bytes
        goff[j]   = (uint64_t)id * 8;                    // dense global bytes
    }
    const uint32_t lds0 = (uint32_t)(uintptr_t)(&Bbuf[0][0]);
    const uint32_t lds1 = (uint32_t)(uintptr_t)(&Bbuf[1][0]);
    const uint64_t Ebase = (uint64_t)(uintptr_t)E;
    const int tbase = chunk * TPC;

#define ISSUE_TILE(T, LBASE)                                                   \
    do {                                                                       \
        uint64_t gb = Ebase + (uint64_t)(tbase + (T)) * (16 * KP * 2);         \
        _Pragma("unroll")                                                      \
        for (int j = 0; j < 5; ++j) {                                          \
            uint32_t la = (LBASE) + ldsoff[j];                                 \
            uint64_t ga = gb + goff[j];                                        \
            asm volatile("global_load_async_to_lds_b64 %0, %1, off"            \
                         :: "v"(la), "v"(ga) : "memory");                      \
        }                                                                      \
    } while (0)

    float m[8], s[8];
#pragma unroll
    for (int r = 0; r < 8; ++r) { m[r] = -INFINITY; s[r] = 0.f; }

    ISSUE_TILE(0, lds0);

    for (int t = 0; t < TPC; ++t) {
        if (t + 1 < TPC) {
            ISSUE_TILE(t + 1, (t & 1) ? lds0 : lds1);
            asm volatile("s_wait_asynccnt 5" ::: "memory"); // tile t resident (in-order)
        } else {
            asm volatile("s_wait_asynccnt 0" ::: "memory");
        }
        __syncthreads();   // all waves' chunks of tile t are in LDS

        const int n0 = (tbase + t) * 16;
        const float bv = bias[n0 + (lane & 15)];  // issued early, consumed after WMMAs

        const _Float16* tile = &Bbuf[t & 1][0];
        const _Float16* fb   = tile + (lane & 15) * BROW + ((lane < 16) ? 0 : 8);

        // Software-pipelined fragment loads: issue kt+1's ds loads before wmma kt
        // so LDS latency overlaps WMMA execution instead of stalling on dscnt 0.
        v8h lo = *(const v8h*)fb;        // ds_load_b128
        v8h hi = *(const v8h*)(fb + 16); // ds_load_b128
        v8f acc = {};
#pragma unroll
        for (int kt = 0; kt < KSTEPS; ++kt) {
            v8h lon = lo, hin = hi;
            if (kt + 1 < KSTEPS) {
                const _Float16* p = fb + (kt + 1) * 32;
                lon = *(const v8h*)p;
                hin = *(const v8h*)(p + 16);
            }
            v16h b;
#pragma unroll
            for (int i = 0; i < 8; ++i) { b[i] = lo[i]; b[i + 8] = hi[i]; }
            acc = __builtin_amdgcn_wmma_f32_16x16x32_f16(
                      false, afrag[kt], false, b, (short)0, acc, false, false);
            lo = lon;
            hi = hin;
        }

#pragma unroll
        for (int r = 0; r < 8; ++r) {
            float v  = acc[r] + bv;  // logit at (row0 + r + 8*(lane>=16), n0 + lane&15)
            float nm = fmaxf(m[r], v);
            s[r] = s[r] * __expf(m[r] - nm) + __expf(v - nm);
            m[r] = nm;
        }
        __syncthreads();   // all reads of buf[t&1] done before it is refilled
    }
#undef ISSUE_TILE

    // Reduce each row's (m,s) across its 16 lanes (xor stays within halves).
#pragma unroll
    for (int r = 0; r < 8; ++r) {
#pragma unroll
        for (int mask = 1; mask <= 8; mask <<= 1) {
            float om = __shfl_xor(m[r], mask, 32);
            float os = __shfl_xor(s[r], mask, 32);
            float nm = fmaxf(m[r], om);
            s[r] = s[r] * __expf(m[r] - nm) + os * __expf(om - nm);
            m[r] = nm;
        }
    }

    // Waves own disjoint rows: write 16 partials directly (lane 0 / lane 16).
    if ((lane & 15) == 0) {
        int half = lane >> 4;   // 0: rows 0..7, 1: rows 8..15
#pragma unroll
        for (int r = 0; r < 8; ++r)
            partials[(size_t)(row0 + half * 8 + r) * NCHUNKS + chunk] =
                make_float2(m[r], s[r]);
    }
}

// ---------------------------------------------------------------------------
// Combine chunk partials -> LSE[b] = m + log(s)
// ---------------------------------------------------------------------------
__global__ void lse_reduce_kernel(const float2* __restrict__ partials,
                                  float* __restrict__ lse) {
    int b = blockIdx.x * blockDim.x + threadIdx.x;
    if (b >= BATCH) return;
    float M = -INFINITY, S = 0.f;
    for (int c = 0; c < NCHUNKS; ++c) {
        float2 p  = partials[(size_t)b * NCHUNKS + c];
        float  nm = fmaxf(M, p.x);
        S = S * __expf(M - nm) + p.y * __expf(p.x - nm);
        M = nm;
    }
    lse[b] = M + __logf(S);
}

// ---------------------------------------------------------------------------
// Token NLL: one block per batch row; 128 threads = 64 X-positions + 64 Y.
// nll[b] = sum over non-pad tokens of (LSE - (dot + bias[tok]))
// ---------------------------------------------------------------------------
__global__ __launch_bounds__(128)
void token_nll_kernel(const _Float16* __restrict__ A,
                      const _Float16* __restrict__ Ex,
                      const _Float16* __restrict__ Ey,
                      const int* __restrict__ tokx, const int* __restrict__ toky,
                      const float* __restrict__ biasx, const float* __restrict__ biasy,
                      const float* __restrict__ lsex,  const float* __restrict__ lsey,
                      float* __restrict__ nll) {
    const int b    = blockIdx.x;
    const int t    = threadIdx.x;
    const int side = t >> 6;
    const int pos  = t & 63;

    const int*      tok  = side ? toky  : tokx;
    const _Float16* E    = side ? Ey    : Ex;
    const float*    bias = side ? biasy : biasx;
    const float*    lse  = side ? lsey  : lsex;

    const int token = tok[b * LEN_TOK + pos];
    float v = 0.f;
    if (token != 0) {
        const _Float16* a = A + (size_t)b * KP;
        const _Float16* e = E + (size_t)token * KP;
        float dot = 0.f;
#pragma unroll 4
        for (int k = 0; k < KP; k += 8) {
            v8h av = *(const v8h*)(a + k);
            v8h ev = *(const v8h*)(e + k);
#pragma unroll
            for (int i = 0; i < 8; ++i) dot += (float)av[i] * (float)ev[i];
        }
        v = lse[b] - (dot + bias[token]);
    }
    __shared__ float red[128];
    red[t] = v;
    __syncthreads();
    for (int st = 64; st > 0; st >>= 1) {
        if (t < st) red[t] += red[t + st];
        __syncthreads();
    }
    if (t == 0) nll[b] = red[0];
}

// ---------------------------------------------------------------------------
// Final mean over the batch.
// ---------------------------------------------------------------------------
__global__ __launch_bounds__(256)
void mean_kernel(const float* __restrict__ nll, float* __restrict__ out) {
    __shared__ float red[256];
    int t = threadIdx.x;
    float v = 0.f;
    for (int i = t; i < BATCH; i += 256) v += nll[i];
    red[t] = v;
    __syncthreads();
    for (int st = 128; st > 0; st >>= 1) {
        if (t < st) red[t] += red[t + st];
        __syncthreads();
    }
    if (t == 0) out[0] = red[0] * (1.f / (float)BATCH);
}

// ---------------------------------------------------------------------------
// Launch
// ---------------------------------------------------------------------------
extern "C" void kernel_launch(void* const* d_in, const int* in_sizes, int n_in,
                              void* d_out, int out_size, void* d_ws, size_t ws_size,
                              hipStream_t stream) {
    const float* samp_hid = (const float*)d_in[0]; // [1024, 300]
    const int*   batch_x  = (const int*)  d_in[1]; // [1024, 64]
    const int*   batch_y  = (const int*)  d_in[2]; // [1024, 64]
    const float* emb_x    = (const float*)d_in[3]; // [50000, 300]
    const float* emb_y    = (const float*)d_in[4]; // [50000, 300]
    const float* bias_x   = (const float*)d_in[5]; // [50000]
    const float* bias_y   = (const float*)d_in[6]; // [50000]
    float*       out      = (float*)d_out;

    // Workspace layout (all offsets multiples of 256 bytes)
    char* ws = (char*)d_ws;
    _Float16* Af16 = (_Float16*)(ws + 0);                          //   655,360 B
    _Float16* Ex16 = (_Float16*)(ws + 655360);                     // 32,000,000 B
    _Float16* Ey16 = (_Float16*)(ws + 32655360);                   // 32,000,000 B
    float2*   Px   = (float2*)  (ws + 64655360);                   //  1,024,000 B
    float2*   Py   = (float2*)  (ws + 65679360);                   //  1,024,000 B
    float*    LSEx = (float*)   (ws + 66703360);                   //      4,096 B
    float*    LSEy = (float*)   (ws + 66707456);                   //      4,096 B
    float*    NLL  = (float*)   (ws + 66711552);                   //      4,096 B

    // 1) f32 -> f16 with K padding (A once, each embedding table once).
    {
        size_t nA = (size_t)BATCH * KP;
        cvt_pad_kernel<<<dim3((unsigned)((nA + 255) / 256)), 256, 0, stream>>>(
            samp_hid, Af16, BATCH);
        size_t nE = (size_t)VOCAB * KP;
        unsigned gE = (unsigned)((nE + 255) / 256);
        cvt_pad_kernel<<<dim3(gE), 256, 0, stream>>>(emb_x, Ex16, VOCAB);
        cvt_pad_kernel<<<dim3(gE), 256, 0, stream>>>(emb_y, Ey16, VOCAB);
    }

    // 2) WMMA GEMM fused with online softmax partials (per side).
    {
        dim3 grid(NCHUNKS, MGROUPS);
        lse_gemm_kernel<<<grid, 256, 0, stream>>>(Af16, Ex16, bias_x, Px);
        lse_gemm_kernel<<<grid, 256, 0, stream>>>(Af16, Ey16, bias_y, Py);
    }

    // 3) Combine partials into LSE per row, per side.
    lse_reduce_kernel<<<dim3(BATCH / 256), 256, 0, stream>>>(Px, LSEx);
    lse_reduce_kernel<<<dim3(BATCH / 256), 256, 0, stream>>>(Py, LSEy);

    // 4) Token gather + per-row NLL.
    token_nll_kernel<<<dim3(BATCH), 128, 0, stream>>>(
        Af16, Ex16, Ey16, batch_x, batch_y, bias_x, bias_y, LSEx, LSEy, NLL);

    // 5) Mean over batch -> scalar output.
    mean_kernel<<<dim3(1), 256, 0, stream>>>(NLL, out);
}
